// BahdanauAttention_19026705121347
// MI455X (gfx1250) — matrix operand order repaired and checked by reading, not compiled, not run on hardware
//
#include <hip/hip_runtime.h>

typedef __attribute__((ext_vector_type(16))) _Float16 v16h;
typedef __attribute__((ext_vector_type(8)))  _Float16 v8h;
typedef __attribute__((ext_vector_type(8)))  float    v8f;
typedef __attribute__((ext_vector_type(4)))  float    v4f;

#define SEQ   1024
#define BATCH 64
#define H1    1024
#define ATT   1024
#define H2    1024
#define DEC_H 512

// LDS row stride in halves: +8 pad rotates banks by 4 per row -> conflict-free b128 reads
#define ASTRIDE (H1 + 8)   // 1032 halves = 2064 bytes (16B aligned)

// ---------------------------------------------------------------------------
// Stage 0: q[b][a] = dot(hs_b[b], W2_w[a]) + W1_b[a] + W2_b[a]
// hs is (L=2, B, Dh=512); hs_b[b] flat index h -> hs[(h>>9)*B*Dh + b*Dh + (h&511)]
// ---------------------------------------------------------------------------
__global__ void __launch_bounds__(256)
q_kernel(const float* __restrict__ hs, const float* __restrict__ W2,
         const float* __restrict__ W1b, const float* __restrict__ W2b,
         float* __restrict__ qout) {
  __shared__ __align__(16) float hsm[H2];
  const int b = blockIdx.x, tid = threadIdx.x;
#pragma unroll
  for (int k = 0; k < H2 / 256; ++k) {
    int h = tid + k * 256;
    int l = h >> 9, d = h & 511;
    hsm[h] = hs[((size_t)l * BATCH + b) * DEC_H + d];
  }
  __syncthreads();
  for (int rep = 0; rep < ATT / 256; ++rep) {
    int a = rep * 256 + tid;
    const v4f* wrow = (const v4f*)(W2 + (size_t)a * H2);
    const v4f* hv   = (const v4f*)hsm;
    v4f acc4 = {0.f, 0.f, 0.f, 0.f};
    for (int i = 0; i < H2 / 4; ++i) acc4 += wrow[i] * hv[i];
    qout[b * ATT + a] = acc4[0] + acc4[1] + acc4[2] + acc4[3] + W1b[a] + W2b[a];
  }
}

// ---------------------------------------------------------------------------
// Stage 1: fused energy GEMM + score.
// Block = one s (64 rows r = s*64+b, contiguous in ht). 512 threads = 16 waves.
// Wave w: mtile = w&3 (16 b-rows), nslot = w>>2 (16 a-cols of the 64-wide A-tile).
// score[s][b] = sum_a V[a]*tanh( ht[s][b]. W1[a] + q[b][a] ) + V_b
// ---------------------------------------------------------------------------
__global__ void __launch_bounds__(512)
energy_score_kernel(const float* __restrict__ ht, const float* __restrict__ W1,
                    const float* __restrict__ q,  const float* __restrict__ Vw,
                    const float* __restrict__ Vb, float* __restrict__ scores) {
  extern __shared__ char smem[];
  _Float16* sA = (_Float16*)smem;               // 64 x ASTRIDE f16 (ht block)
  _Float16* sB = sA + 64 * ASTRIDE;             // 64 x ASTRIDE f16 (W1 tile)
  float*    sc = (float*)(sB + 64 * ASTRIDE);   // 16 x 64 score partials

  const int tid   = threadIdx.x;
  const int lane  = tid & 31;
  const int wave  = tid >> 5;
  const int mtile = wave & 3;
  const int nslot = wave >> 2;
  const int s     = blockIdx.x;

  // zero the per-wave score scratch (16*64 = 1024 floats)
  sc[tid] = 0.f;
  sc[tid + 512] = 0.f;

  // stage ht block (one s, all 64 b, contiguous 256 KB) -> f16 LDS, read ONCE per block
  const float* htblk = ht + (size_t)s * BATCH * H1;
  for (int i = tid; i < (BATCH * H1) / 4; i += 512) {
    v4f v = ((const v4f*)htblk)[i];
    int row = (i * 4) / H1, k = (i * 4) % H1;
    _Float16* dst = sA + row * ASTRIDE + k;
    dst[0] = (_Float16)v[0]; dst[1] = (_Float16)v[1];
    dst[2] = (_Float16)v[2]; dst[3] = (_Float16)v[3];
  }

  float svec[8];
#pragma unroll
  for (int j = 0; j < 8; ++j) svec[j] = 0.f;

  const int l15   = lane & 15;   // A-row (M) within tile / B-col (N) within tile
  const int ahalf = lane >> 4;   // K-half selector per ISA lane layout

  // ISA 7.12.2 fragment base pointers:
  //   A lane<16: K 0..7 then 16..23 ; lane>=16: K 8..15 then 24..31  -> +ahalf*8
  //   B lane<16: K 0..7 then 16..23 ; lane>=16: K 8..15 then 24..31  -> +ahalf*8
  const _Float16* arow_p = sA + (mtile * 16 + l15) * ASTRIDE + ahalf * 8;
  const _Float16* brow_p = sB + (nslot * 16 + l15) * ASTRIDE + ahalf * 8;

  for (int it = 0; it < ATT / 64; ++it) {
    const int abase = it * 64;
    __syncthreads();  // prev iteration's consumers done; also covers sA/sc init on it=0
    const float* w1blk = W1 + (size_t)abase * H1;   // 64 W1 rows (L2/WGP$-resident)
    for (int i = tid; i < (64 * H1) / 4; i += 512) {
      v4f v = ((const v4f*)w1blk)[i];
      int row = (i * 4) / H1, k = (i * 4) % H1;
      _Float16* dst = sB + row * ASTRIDE + k;
      dst[0] = (_Float16)v[0]; dst[1] = (_Float16)v[1];
      dst[2] = (_Float16)v[2]; dst[3] = (_Float16)v[3];
    }
    __syncthreads();

    // Prefetch the NEXT W1 tile (256 KB = 4 cachelines/thread) into WGP$ so its
    // staging loads hit close caches; overlaps with the WMMA work below.
    if (it + 1 < ATT / 64) {
      const char* nxt = (const char*)(W1 + (size_t)(abase + 64) * H1);
#pragma unroll
      for (int p = 0; p < 4; ++p)
        __builtin_prefetch(nxt + (size_t)tid * 512 + p * 128, 0, 3);
    }

    v8f c = {};
#pragma unroll
    for (int k0 = 0; k0 < H1; k0 += 32) {
      v8h alo = *(const v8h*)(arow_p + k0);
      v8h ahi = *(const v8h*)(arow_p + k0 + 16);
      v8h blo = *(const v8h*)(brow_p + k0);
      v8h bhi = *(const v8h*)(brow_p + k0 + 16);
      v16h af, bf;
#pragma unroll
      for (int e = 0; e < 8; ++e) {
        af[e] = alo[e]; af[e + 8] = ahi[e];
        bf[e] = blo[e]; bf[e + 8] = bhi[e];
      }
      c = __builtin_amdgcn_wmma_f32_16x16x32_f16(
              false, af, false, bf, (short)0, c, false, false);
    }

    // epilogue: +q, tanh, *V[a] -> accumulate score (C layout: elem j = row ahalf*8+j, col l15)
    const int a  = abase + nslot * 16 + l15;
    const float vwa = Vw[a];
    const int b0 = mtile * 16 + ahalf * 8;
#pragma unroll
    for (int j = 0; j < 8; ++j) {
      float x = c[j] + q[(b0 + j) * ATT + a];
      svec[j] += vwa * tanhf(x);
    }
  }

  // reduce each row's score across the 16 lanes holding it (fixed-order butterfly)
#pragma unroll
  for (int j = 0; j < 8; ++j) {
    float v = svec[j];
    v += __shfl_xor(v, 1, 32);
    v += __shfl_xor(v, 2, 32);
    v += __shfl_xor(v, 4, 32);
    v += __shfl_xor(v, 8, 32);
    if ((lane & 15) == 0)
      sc[wave * 64 + mtile * 16 + ahalf * 8 + j] = v;
  }
  __syncthreads();
  if (tid < 64) {
    float acc = Vb[0];
#pragma unroll
    for (int w = 0; w < 16; ++w) acc += sc[w * 64 + tid];
    scores[s * BATCH + tid] = acc;   // scores layout [S][B]
  }
}

// ---------------------------------------------------------------------------
// Stage 2: softmax over s for each b; write weights to d_out (B,S) layout
// ---------------------------------------------------------------------------
__global__ void __launch_bounds__(256)
softmax_kernel(const float* __restrict__ scores, float* __restrict__ wout) {
  __shared__ float red[8];
  const int b = blockIdx.x, tid = threadIdx.x;
  float v[4];
#pragma unroll
  for (int k = 0; k < 4; ++k) v[k] = scores[(tid + k * 256) * BATCH + b];
  float m = fmaxf(fmaxf(v[0], v[1]), fmaxf(v[2], v[3]));
#pragma unroll
  for (int o = 16; o >= 1; o >>= 1) m = fmaxf(m, __shfl_xor(m, o, 32));
  if ((tid & 31) == 0) red[tid >> 5] = m;
  __syncthreads();
  float bm = red[0];
#pragma unroll
  for (int w = 1; w < 8; ++w) bm = fmaxf(bm, red[w]);
  __syncthreads();
  float e[4], sum = 0.f;
#pragma unroll
  for (int k = 0; k < 4; ++k) { e[k] = __expf(v[k] - bm); sum += e[k]; }
#pragma unroll
  for (int o = 16; o >= 1; o >>= 1) sum += __shfl_xor(sum, o, 32);
  if ((tid & 31) == 0) red[tid >> 5] = sum;
  __syncthreads();
  float bs = 0.f;
#pragma unroll
  for (int w = 0; w < 8; ++w) bs += red[w];
  float inv = 1.0f / bs;
#pragma unroll
  for (int k = 0; k < 4; ++k) wout[b * SEQ + tid + k * 256] = e[k] * inv;
}

// ---------------------------------------------------------------------------
// Stage 3: context partials: part[chunk][b][h] = sum_{s in chunk} w[b][s]*ht[s][b][h]
// ---------------------------------------------------------------------------
__global__ void __launch_bounds__(256)
context_partial_kernel(const float* __restrict__ ht, const float* __restrict__ w,
                       float* __restrict__ part) {
  __shared__ float wl[64];
  const int b = blockIdx.x, chunk = blockIdx.y, tid = threadIdx.x;
  const int s0 = chunk * 64;
  if (tid < 64) wl[tid] = w[b * SEQ + s0 + tid];
  __syncthreads();
  v4f acc = {0.f, 0.f, 0.f, 0.f};
  for (int i = 0; i < 64; ++i) {
    v4f hv = *(const v4f*)(ht + ((size_t)(s0 + i) * BATCH + b) * H1 + tid * 4);
    acc += hv * wl[i];
  }
  *(v4f*)(part + ((size_t)(chunk * BATCH + b)) * H1 + tid * 4) = acc;
}

// Stage 4: deterministic reduce of the 16 chunks -> context (B,H1) at d_out[0]
__global__ void __launch_bounds__(256)
context_reduce_kernel(const float* __restrict__ part, float* __restrict__ ctx) {
  const int b = blockIdx.x, tid = threadIdx.x;
  v4f acc = {0.f, 0.f, 0.f, 0.f};
#pragma unroll
  for (int c = 0; c < 16; ++c)
    acc += *(const v4f*)(part + ((size_t)(c * BATCH + b)) * H1 + tid * 4);
  *(v4f*)(ctx + (size_t)b * H1 + tid * 4) = acc;
}

// ---------------------------------------------------------------------------
extern "C" void kernel_launch(void* const* d_in, const int* in_sizes, int n_in,
                              void* d_out, int out_size, void* d_ws, size_t ws_size,
                              hipStream_t stream) {
  (void)in_sizes; (void)n_in; (void)out_size; (void)ws_size;
  const float* hs  = (const float*)d_in[0];
  const float* ht  = (const float*)d_in[1];
  const float* W1w = (const float*)d_in[2];
  const float* W1b = (const float*)d_in[3];
  const float* W2w = (const float*)d_in[4];
  const float* W2b = (const float*)d_in[5];
  const float* Vw  = (const float*)d_in[6];
  const float* Vb  = (const float*)d_in[7];

  float* out = (float*)d_out;
  float* ctx = out;                   // (B, H1)
  float* wts = out + BATCH * H1;      // (B, S, 1)

  float* ws     = (float*)d_ws;
  float* qbuf   = ws;                      // B*ATT          = 65536 f
  float* scores = qbuf + BATCH * ATT;      // S*B            = 65536 f
  float* part   = scores + SEQ * BATCH;    // 16*B*H1        = 1 Mi f  (~4.5 MB total)

  q_kernel<<<BATCH, 256, 0, stream>>>(hs, W2w, W1b, W2b, qbuf);

  const size_t lds_bytes = (size_t)(2 * 64 * ASTRIDE) * sizeof(_Float16)
                         + (size_t)(16 * 64) * sizeof(float);   // ~268 KB of 320 KB WGP LDS
  (void)hipFuncSetAttribute((const void*)energy_score_kernel,
                            hipFuncAttributeMaxDynamicSharedMemorySize,
                            (int)lds_bytes);
  energy_score_kernel<<<SEQ, 512, lds_bytes, stream>>>(ht, W1w, qbuf, Vw, Vb, scores);

  softmax_kernel<<<BATCH, 256, 0, stream>>>(scores, wts);
  context_partial_kernel<<<dim3(BATCH, 16), 256, 0, stream>>>(ht, wts, part);
  context_reduce_kernel<<<BATCH, 256, 0, stream>>>(part, ctx);
}
